// PointerNetwork_18064632447613
// MI455X (gfx1250) — compile-verified, matrix operand order checked by reading, not run on hardware
//
#include <hip/hip_runtime.h>
#include <hip/hip_bf16.h>

typedef _Float16 v16h __attribute__((ext_vector_type(16)));
typedef _Float16 v8h  __attribute__((ext_vector_type(8)));
typedef float    v8f  __attribute__((ext_vector_type(8)));

#define HPAD 520   // 512 + 8 halves pad -> row stride 1040B, kills LDS bank conflicts

__device__ __forceinline__ float sigf(float x) {
    x = fminf(fmaxf(x, -30.f), 30.f);
    return __builtin_amdgcn_rcpf(1.f + __expf(-x));
}
__device__ __forceinline__ float tanh_fast(float x) {
    x = fminf(fmaxf(x, -15.f), 15.f);
    float e = __expf(2.f * x);
    return (e - 1.f) * __builtin_amdgcn_rcpf(e + 1.f);
}

// ---------------- grid sync (persistent decoder) ----------------
__device__ __forceinline__ void gsync(unsigned* cnt, unsigned* gen, unsigned nb) {
    __threadfence();
    __syncthreads();
    if (threadIdx.x == 0) {
        unsigned g = __hip_atomic_load(gen, __ATOMIC_ACQUIRE, __HIP_MEMORY_SCOPE_AGENT);
        unsigned a = __hip_atomic_fetch_add(cnt, 1u, __ATOMIC_ACQ_REL, __HIP_MEMORY_SCOPE_AGENT);
        if (a == nb - 1u) {
            __hip_atomic_store(cnt, 0u, __ATOMIC_RELAXED, __HIP_MEMORY_SCOPE_AGENT);
            __hip_atomic_fetch_add(gen, 1u, __ATOMIC_RELEASE, __HIP_MEMORY_SCOPE_AGENT);
        } else {
            while (__hip_atomic_load(gen, __ATOMIC_ACQUIRE, __HIP_MEMORY_SCOPE_AGENT) == g)
                __builtin_amdgcn_s_sleep(2);
        }
    }
    __syncthreads();
}

// ---------------- init: zero sync counters ----------------
__global__ void init_kernel(unsigned* sync) {
    if (threadIdx.x < 2) sync[threadIdx.x] = 0u;
}

// ---------------- input projection: x[s][b][h] = inputs[b][s][:]@Wp.T + bp ----------------
__global__ void proj_kernel(const float* __restrict__ inp, const float* __restrict__ Wp,
                            const float* __restrict__ bp, float* __restrict__ x) {
    int tid = blockIdx.x * blockDim.x + threadIdx.x;
    if (tid >= 128 * 32 * 512) return;
    int h = tid & 511, b = (tid >> 9) & 31, s = tid >> 14;
    float i0 = inp[((size_t)b * 128 + s) * 2 + 0];
    float i1 = inp[((size_t)b * 128 + s) * 2 + 1];
    x[tid] = i0 * Wp[h * 2 + 0] + i1 * Wp[h * 2 + 1] + bp[h];
}

// ---------------- repack fp32 weight (N x K, row major) into WMMA B-operand f16 layout ----
// packed[((ntile*chunksTotal + chunkOff + ch)*32 + lane)*16 + j]
// lane: n = ntile*16 + (lane&15); k(j) = ch*32 + (j/8)*16 + (j%8) + (lane>=16 ? 8 : 0)
__global__ void repack_kernel(const float* __restrict__ W, _Float16* __restrict__ out,
                              int N, int K, int chunksTotal, int chunkOff) {
    int tid = blockIdx.x * blockDim.x + threadIdx.x;
    int lane = tid & 31, idx = tid >> 5;
    int nch = K >> 5;
    if (idx >= (N >> 4) * nch) return;
    int ch = idx % nch, ntile = idx / nch;
    int n = ntile * 16 + (lane & 15);
    int kb = ch * 32 + ((lane >> 4) << 3);
    const float* src = W + (size_t)n * K + kb;
    _Float16* dst = out + ((size_t)((ntile * chunksTotal + chunkOff + ch) * 32) + lane) * 16;
#pragma unroll
    for (int j = 0; j < 8; ++j) { dst[j] = (_Float16)src[j]; dst[8 + j] = (_Float16)src[16 + j]; }
}

// ---------------- generic parallel WMMA GEMM: Out[m][n] = A[m][:]·W[n][:] + bias ----------
__global__ __launch_bounds__(256) void gemm_ws(const float* __restrict__ A,
                                               const _Float16* __restrict__ Bp,
                                               float* __restrict__ Out,
                                               const float* __restrict__ bias0,
                                               const float* __restrict__ bias1,
                                               int M, int N, int K) {
    int ntn = N >> 4, nch = K >> 5;
    int lane = threadIdx.x & 31, wv = threadIdx.x >> 5;
    int wpb = blockDim.x >> 5;
    int gw = blockIdx.x * wpb + wv, gws = gridDim.x * wpb;
    int ngrp_n = ntn >> 3;                       // 8 n-tiles per group (A reuse)
    int ngroups = (M >> 4) * ngrp_n;
    int mrow = lane & 15, hi = lane >> 4, koff = hi * 8;
    for (int grp = gw; grp < ngroups; grp += gws) {
        int mt = grp / ngrp_n, ng = grp % ngrp_n;
        v8f acc[8] = {};
        for (int ch = 0; ch < nch; ++ch) {
            const float* ar = A + (size_t)(mt * 16 + mrow) * K + ch * 32 + koff;
            v16h a;
#pragma unroll
            for (int j = 0; j < 8; ++j) { a[j] = (_Float16)ar[j]; a[j + 8] = (_Float16)ar[j + 16]; }
#pragma unroll
            for (int q = 0; q < 8; ++q) {
                int nt = ng * 8 + q;
                v16h bm = ((const v16h*)Bp)[(size_t)(nt * nch + ch) * 32 + lane];
                acc[q] = __builtin_amdgcn_wmma_f32_16x16x32_f16(false, a, false, bm,
                                                                (short)0, acc[q], false, false);
            }
        }
#pragma unroll
        for (int q = 0; q < 8; ++q) {
            int nt = ng * 8 + q;
            int ncol = nt * 16 + (lane & 15);
            float bs = (bias0 ? bias0[ncol] : 0.f) + (bias1 ? bias1[ncol] : 0.f);
#pragma unroll
            for (int r = 0; r < 8; ++r) {
                int m = mt * 16 + r + hi * 8;
                Out[(size_t)m * N + ncol] = acc[q][r] + bs;
            }
        }
    }
}

// ---------------- persistent encoder: 128 LSTM steps in one workgroup -------------------
__global__ __launch_bounds__(1024) void encoder_kernel(const _Float16* __restrict__ pWhh,
                                                       const float* __restrict__ xg,
                                                       float* __restrict__ enc_out,
                                                       float* __restrict__ h_state,
                                                       float* __restrict__ c_state) {
    __shared__ _Float16 hA[32][HPAD];            // h in f16, WMMA A source
    int tid = threadIdx.x, lane = tid & 31, wv = tid >> 5;
    int mrow = lane & 15, hi = lane >> 4, koff = hi * 8;
    int ut = wv;                                 // 32 waves <-> 32 u-tiles; pairs p=mt(0,1)
    for (int i = tid; i < 32 * HPAD; i += 1024) ((_Float16*)hA)[i] = (_Float16)0.f;
    v8f cc[2] = {};                              // cell state, register-resident (D layout)
    __syncthreads();
    for (int s = 0; s < 128; ++s) {
        v8f acc[2][4] = {};
        for (int ch = 0; ch < 16; ++ch) {
            v16h bm[4];
#pragma unroll
            for (int g = 0; g < 4; ++g)
                bm[g] = ((const v16h*)pWhh)[(size_t)((g * 32 + ut) * 16 + ch) * 32 + lane];
#pragma unroll
            for (int p = 0; p < 2; ++p) {
                const _Float16* ar = &hA[p * 16 + mrow][ch * 32 + koff];
                v8h lo = *(const v8h*)ar, hp = *(const v8h*)(ar + 16);
                v16h a = __builtin_shufflevector(lo, hp, 0, 1, 2, 3, 4, 5, 6, 7,
                                                 8, 9, 10, 11, 12, 13, 14, 15);
#pragma unroll
                for (int g = 0; g < 4; ++g)
                    acc[p][g] = __builtin_amdgcn_wmma_f32_16x16x32_f16(false, a, false, bm[g],
                                                                       (short)0, acc[p][g], false, false);
            }
        }
        __syncthreads();                          // all reads of old h done
        int u = ut * 16 + (lane & 15);
#pragma unroll
        for (int p = 0; p < 2; ++p) {
#pragma unroll
            for (int r = 0; r < 8; ++r) {
                int b = p * 16 + r + hi * 8;
                const float* xr = xg + ((size_t)(s * 32 + b) * 2048 + u);
                float gi = acc[p][0][r] + xr[0];
                float gf = acc[p][1][r] + xr[512];
                float gg = acc[p][2][r] + xr[1024];
                float go = acc[p][3][r] + xr[1536];
                float fi = sigf(gi), ff = sigf(gf), fg = tanh_fast(gg), fo = sigf(go);
                float cn = ff * cc[p][r] + fi * fg;
                cc[p][r] = cn;
                float h = fo * tanh_fast(cn);
                hA[b][u] = (_Float16)h;
                enc_out[((size_t)b * 128 + s) * 512 + u] = h;
                if (s == 127) { h_state[b * 512 + u] = h; c_state[b * 512 + u] = cn; }
            }
        }
        __syncthreads();                          // new h visible before next step
    }
}

// ---------------- persistent decoder: 32 blocks, redundant LSTM, split attention --------
__global__ __launch_bounds__(1024) void decoder_kernel(
    const _Float16* __restrict__ pDec, const _Float16* __restrict__ pW2,
    const float* __restrict__ dbih, const float* __restrict__ dbhh,
    const float* __restrict__ vvec, const float* __restrict__ enc_out,
    const float* __restrict__ enc_proj, const float* __restrict__ h0,
    const float* __restrict__ c0, float* __restrict__ scores_g,
    int* __restrict__ topidx, unsigned* __restrict__ syncmem,
    float* __restrict__ out) {
    __shared__ _Float16 hA[32][HPAD];             // h (f16)
    __shared__ _Float16 buf[32][HPAD];            // dec_in (f16), reused for dec_proj (f16)
    int tid = threadIdx.x, lane = tid & 31, wv = tid >> 5, bx = blockIdx.x;
    int mrow = lane & 15, hi = lane >> 4, koff = hi * 8, ut = wv;
    unsigned* scnt = syncmem; unsigned* sgen = syncmem + 1;

    for (int i = tid; i < 32 * 512; i += 1024) {
        int b = i >> 9, u = i & 511;
        hA[b][u] = (_Float16)h0[i];
        buf[b][u] = (_Float16)0.f;                // dec_in0 = 0
    }
    v8f cc[2];
#pragma unroll
    for (int p = 0; p < 2; ++p)
#pragma unroll
        for (int r = 0; r < 8; ++r)
            cc[p][r] = c0[(size_t)(p * 16 + r + hi * 8) * 512 + ut * 16 + (lane & 15)];
    __syncthreads();

    for (int t = 0; t < 128; ++t) {
        // --- fused gates GEMM: [dec_in | h] (32x1024) x packed [Wih;Whh] -> 32x2048 ---
        v8f acc[2][4] = {};
        for (int ch = 0; ch < 32; ++ch) {
            v16h bm[4];
#pragma unroll
            for (int g = 0; g < 4; ++g)
                bm[g] = ((const v16h*)pDec)[(size_t)((g * 32 + ut) * 32 + ch) * 32 + lane];
#pragma unroll
            for (int p = 0; p < 2; ++p) {
                const _Float16* ar = (ch < 16) ? &buf[p * 16 + mrow][ch * 32 + koff]
                                               : &hA[p * 16 + mrow][(ch - 16) * 32 + koff];
                v8h lo = *(const v8h*)ar, hp = *(const v8h*)(ar + 16);
                v16h a = __builtin_shufflevector(lo, hp, 0, 1, 2, 3, 4, 5, 6, 7,
                                                 8, 9, 10, 11, 12, 13, 14, 15);
#pragma unroll
                for (int g = 0; g < 4; ++g)
                    acc[p][g] = __builtin_amdgcn_wmma_f32_16x16x32_f16(false, a, false, bm[g],
                                                                       (short)0, acc[p][g], false, false);
            }
        }
        __syncthreads();
        int u = ut * 16 + (lane & 15);
#pragma unroll
        for (int p = 0; p < 2; ++p) {
#pragma unroll
            for (int r = 0; r < 8; ++r) {
                int b = p * 16 + r + hi * 8;
                float gi = acc[p][0][r] + dbih[u] + dbhh[u];
                float gf = acc[p][1][r] + dbih[512 + u] + dbhh[512 + u];
                float gg = acc[p][2][r] + dbih[1024 + u] + dbhh[1024 + u];
                float go = acc[p][3][r] + dbih[1536 + u] + dbhh[1536 + u];
                float fi = sigf(gi), ff = sigf(gf), fg = tanh_fast(gg), fo = sigf(go);
                float cn = ff * cc[p][r] + fi * fg;
                cc[p][r] = cn;
                hA[b][u] = (_Float16)(fo * tanh_fast(cn));
            }
        }
        __syncthreads();
        // --- dec_proj = h @ W2.T (32x512), write into buf as f16 ---
        v8f pacc[2] = {};
        for (int ch = 0; ch < 16; ++ch) {
            v16h bm = ((const v16h*)pW2)[(size_t)(wv * 16 + ch) * 32 + lane];
#pragma unroll
            for (int p = 0; p < 2; ++p) {
                const _Float16* ar = &hA[p * 16 + mrow][ch * 32 + koff];
                v8h lo = *(const v8h*)ar, hp = *(const v8h*)(ar + 16);
                v16h a = __builtin_shufflevector(lo, hp, 0, 1, 2, 3, 4, 5, 6, 7,
                                                 8, 9, 10, 11, 12, 13, 14, 15);
                pacc[p] = __builtin_amdgcn_wmma_f32_16x16x32_f16(false, a, false, bm,
                                                                 (short)0, pacc[p], false, false);
            }
        }
#pragma unroll
        for (int p = 0; p < 2; ++p)
#pragma unroll
            for (int r = 0; r < 8; ++r)
                buf[p * 16 + r + hi * 8][wv * 16 + (lane & 15)] = (_Float16)pacc[p][r];
        __syncthreads();
        // --- attention scores for this block's s-slice: s = bx*4 .. bx*4+3 ---
        {
            int b = wv;
#pragma unroll
            for (int q = 0; q < 4; ++q) {
                int s = bx * 4 + q;
                const float* ep = enc_proj + ((size_t)(b * 128 + s) * 512) + lane * 16;
                const _Float16* dp = &buf[b][lane * 16];
                float a = 0.f;
#pragma unroll
                for (int j = 0; j < 16; ++j)
                    a += vvec[lane * 16 + j] * tanh_fast(ep[j] + (float)dp[j]);
#pragma unroll
                for (int off = 16; off > 0; off >>= 1) a += __shfl_xor(a, off, 32);
                if (lane == 0) scores_g[b * 128 + s] = a;
            }
        }
        gsync(scnt, sgen, gridDim.x);
        // --- log_softmax + argmax for row b = bx (wave 0) ---
        if (wv == 0) {
            int b = bx;
            float v4[4]; float m = -3.0e38f;
#pragma unroll
            for (int k = 0; k < 4; ++k) { v4[k] = scores_g[b * 128 + lane + 32 * k]; m = fmaxf(m, v4[k]); }
#pragma unroll
            for (int off = 16; off > 0; off >>= 1) m = fmaxf(m, __shfl_xor(m, off, 32));
            float se = 0.f;
#pragma unroll
            for (int k = 0; k < 4; ++k) se += __expf(v4[k] - m);
#pragma unroll
            for (int off = 16; off > 0; off >>= 1) se += __shfl_xor(se, off, 32);
            float lse = __logf(se);
#pragma unroll
            for (int k = 0; k < 4; ++k)
                out[((size_t)b * 128 + t) * 128 + lane + 32 * k] = v4[k] - m - lse;
            float bmv = v4[0]; int bi = lane;
#pragma unroll
            for (int k = 1; k < 4; ++k) if (v4[k] > bmv) { bmv = v4[k]; bi = lane + 32 * k; }
#pragma unroll
            for (int off = 16; off > 0; off >>= 1) {
                float om = __shfl_xor(bmv, off, 32); int oi = __shfl_xor(bi, off, 32);
                if (om > bmv || (om == bmv && oi < bi)) { bmv = om; bi = oi; }
            }
            if (lane == 0) topidx[b] = bi;
        }
        gsync(scnt, sgen, gridDim.x);
        // --- next_in = enc_out[b, top[b]] -> buf (f16) ---
        for (int i = tid; i < 32 * 512; i += 1024) {
            int b = i >> 9, uu = i & 511;
            int s = topidx[b];
            buf[b][uu] = (_Float16)enc_out[((size_t)(b * 128 + s)) * 512 + uu];
        }
        __syncthreads();
    }
}

// ---------------- host-side orchestration ----------------
extern "C" void kernel_launch(void* const* d_in, const int* in_sizes, int n_in,
                              void* d_out, int out_size, void* d_ws, size_t ws_size,
                              hipStream_t stream) {
    const float* inputs  = (const float*)d_in[0];
    const float* Wp      = (const float*)d_in[1];
    const float* bp      = (const float*)d_in[2];
    const float* encWih  = (const float*)d_in[3];
    const float* encWhh  = (const float*)d_in[4];
    const float* encBih  = (const float*)d_in[5];
    const float* encBhh  = (const float*)d_in[6];
    const float* decWih  = (const float*)d_in[7];
    const float* decWhh  = (const float*)d_in[8];
    const float* decBih  = (const float*)d_in[9];
    const float* decBhh  = (const float*)d_in[10];
    const float* W1      = (const float*)d_in[11];
    const float* W2      = (const float*)d_in[12];
    const float* vvec    = (const float*)d_in[13];
    float* outp = (float*)d_out;

    char* ws = (char*)d_ws;
    constexpr size_t SZ_X    = 128ull * 32 * 512 * 4;       // 8 MB  x[s][b][h]
    constexpr size_t SZ_XG   = 128ull * 32 * 2048 * 4;      // 32 MB xg[s][b][4H]
    constexpr size_t SZ_EO   = 32ull * 128 * 512 * 4;       // 8 MB  enc_out[b][s][h]
    constexpr size_t SZ_EP   = SZ_EO;                       // 8 MB  enc_proj
    constexpr size_t SZ_HC   = 32ull * 512 * 4;             // 64 KB
    constexpr size_t SZ_PIH  = 2048ull * 512 * 2;           // 2 MB  packed enc_Wih
    constexpr size_t SZ_PHH  = SZ_PIH;                      // 2 MB  packed enc_Whh
    constexpr size_t SZ_PDEC = 2048ull * 1024 * 2;          // 4 MB  packed [dec_Wih;dec_Whh]
    constexpr size_t SZ_PW   = 512ull * 512 * 2;            // 0.5 MB each W1/W2
    size_t o = 0;
    unsigned* syncmem = (unsigned*)(ws + o); o += 256;
    float* x_ws   = (float*)(ws + o); o += SZ_X;
    float* xg     = (float*)(ws + o); o += SZ_XG;
    float* encOut = (float*)(ws + o); o += SZ_EO;
    float* encPrj = (float*)(ws + o); o += SZ_EP;
    float* hSt    = (float*)(ws + o); o += SZ_HC;
    float* cSt    = (float*)(ws + o); o += SZ_HC;
    int*   topIdx = (int*)(ws + o);   o += 256;
    float* scores = (float*)(ws + o); o += 32ull * 128 * 4;
    _Float16* pEncIh = (_Float16*)(ws + o); o += SZ_PIH;
    _Float16* pEncHh = (_Float16*)(ws + o); o += SZ_PHH;
    _Float16* pDec   = (_Float16*)(ws + o); o += SZ_PDEC;
    _Float16* pW1    = (_Float16*)(ws + o); o += SZ_PW;
    _Float16* pW2    = (_Float16*)(ws + o); o += SZ_PW;

    init_kernel<<<1, 32, 0, stream>>>(syncmem);
    proj_kernel<<<8192, 256, 0, stream>>>(inputs, Wp, bp, x_ws);

    // weight repacks into WMMA B-operand layout (f16)
    repack_kernel<<<256, 256, 0, stream>>>(encWih, pEncIh, 2048, 512, 16, 0);
    repack_kernel<<<256, 256, 0, stream>>>(encWhh, pEncHh, 2048, 512, 16, 0);
    repack_kernel<<<256, 256, 0, stream>>>(decWih, pDec,   2048, 512, 32, 0);   // K-chunks 0..15
    repack_kernel<<<256, 256, 0, stream>>>(decWhh, pDec,   2048, 512, 32, 16);  // K-chunks 16..31
    repack_kernel<<<64, 256, 0, stream>>>(W1, pW1, 512, 512, 16, 0);
    repack_kernel<<<64, 256, 0, stream>>>(W2, pW2, 512, 512, 16, 0);

    // xg = x @ enc_Wih.T + enc_bih + enc_bhh   (M=4096, N=2048, K=512)
    gemm_ws<<<256, 256, 0, stream>>>(x_ws, pEncIh, xg, encBih, encBhh, 4096, 2048, 512);

    // encoder recurrence (single persistent workgroup, 32 waves)
    encoder_kernel<<<1, 1024, 0, stream>>>(pEncHh, xg, encOut, hSt, cSt);

    // enc_proj = enc_out @ W1.T  (M=4096, N=512, K=512)
    gemm_ws<<<256, 256, 0, stream>>>(encOut, pW1, encPrj, nullptr, nullptr, 4096, 512, 512);

    // decoder recurrence: 32 persistent blocks, grid-synced twice per step
    decoder_kernel<<<32, 1024, 0, stream>>>(pDec, pW2, decBih, decBhh, vvec, encOut, encPrj,
                                            hSt, cSt, scores, topIdx, syncmem, outp);

    (void)in_sizes; (void)n_in; (void)out_size; (void)ws_size;
}
